// MBMconv2d_75917841924742
// MI455X (gfx1250) — matrix operand order repaired and checked by reading, not compiled, not run on hardware
//
#include <hip/hip_runtime.h>
#include <hip/hip_bf16.h>

typedef __attribute__((ext_vector_type(16))) _Float16 v16h;
typedef __attribute__((ext_vector_type(8)))  _Float16 v8h;
typedef __attribute__((ext_vector_type(8)))  float    v8f;

#define WAVES 7
#define TPB   (WAVES * 32)       // 224 threads, 7 wave32
#define WROW  584                // f16 per weight row: 576 used + 8 pad (bank spread)
#define XROW  72                 // f16 per (rr,j) row: 64 channels + 8 pad (16B-aligned stride)

// Implicit-GEMM 3x3 SAME conv via V_WMMA_F32_16X16X32_F16.
// One block computes one (n,h) output row: 64 channels x 112 pixels.
__global__ __launch_bounds__(TPB)
void MBMconv2d_75917841924742_kernel(const float* __restrict__ x,
                                     const float* __restrict__ wgt,
                                     float* __restrict__ out) {
  __shared__ __align__(16) _Float16 Wlds[64 * WROW];    // 74,752 B
  __shared__ __align__(16) _Float16 Xlds[3 * 114 * XROW]; // 49,248 B

  const int tid  = threadIdx.x;
  const int wave = tid >> 5;
  const int lane = tid & 31;
  const int blk  = blockIdx.x;
  const int n = blk / 112;
  const int h = blk % 112;

  // ---- stage weights: global [k][c][r][s] -> LDS [k][(r*3+s)*64 + c], f32->f16
  for (int i = tid; i < 64 * 576; i += TPB) {
    int k   = i / 576;
    int rem = i - k * 576;
    int c   = rem / 9;
    int rs  = rem - c * 9;
    Wlds[k * WROW + rs * 64 + c] = (_Float16)wgt[i];
  }

  // ---- stage input rows h-1..h+1, cols -1..112 (zero-padded), all 64 channels
  const float* xb = x + (size_t)n * (64 * 112 * 112);
  for (int i = tid; i < 64 * 342; i += TPB) {
    int c    = i / 342;
    int rem  = i - c * 342;
    int rr   = rem / 114;
    int j    = rem - rr * 114;      // j-1 = global w
    int hr   = h - 1 + rr;
    int wcol = j - 1;
    float v = 0.0f;
    if ((unsigned)hr < 112u && (unsigned)wcol < 112u)
      v = xb[(c * 112 + hr) * 112 + wcol];
    Xlds[(rr * 114 + j) * XROW + c] = (_Float16)v;
  }

  __syncthreads();

  const int hi   = lane >> 4;       // half-wave select
  const int ln   = lane & 15;
  const int wpix = wave * 16 + ln;  // output column 0..111

  v8f acc[4] = {v8f{}, v8f{}, v8f{}, v8f{}};

  for (int kk = 0; kk < 18; ++kk) {
    const int tbase = kk * 32;

    // B fragment: 32x16 f16, lane N = ln, K = t0+e (e=0..15), contiguous in LDS.
    const int t0 = tbase + hi * 16;
    const int rs = t0 >> 6;         // 0..8  (t' = rs*64 + c)
    const int c0 = t0 & 63;         // 0,16,32,48
    const int rr = rs / 3;
    const int s  = rs - rr * 3;
    const _Float16* bp = &Xlds[(rr * 114 + wpix + s) * XROW + c0];
    union { v16h v; v8h q[2]; } b;
    b.q[0] = *(const v8h*)(bp);
    b.q[1] = *(const v8h*)(bp + 8);

    // A fragments per 16-channel M-tile: lane M = ln,
    // K chunk [tbase+hi*16, +16) contiguous.
#pragma unroll
    for (int m = 0; m < 4; ++m) {
      const _Float16* ap = &Wlds[(m * 16 + ln) * WROW + tbase + hi * 16];
      union { v16h v; v8h q[2]; } a;
      a.q[0] = *(const v8h*)(ap);
      a.q[1] = *(const v8h*)(ap + 8);
      acc[m] = __builtin_amdgcn_wmma_f32_16x16x32_f16(
          false, a.v, false, b.v, (short)0, acc[m], false, false);
    }
  }

  // ---- store D: element r of lane -> channel m*16 + hi*8 + r, column wpix
  float* ob = out + (size_t)n * (64 * 112 * 112) + (size_t)h * 112 + wpix;
#pragma unroll
  for (int m = 0; m < 4; ++m) {
#pragma unroll
    for (int r = 0; r < 8; ++r) {
      int kout = m * 16 + hi * 8 + r;
      ob[(size_t)kout * (112 * 112)] = acc[m][r];
    }
  }
}

extern "C" void kernel_launch(void* const* d_in, const int* in_sizes, int n_in,
                              void* d_out, int out_size, void* d_ws, size_t ws_size,
                              hipStream_t stream) {
  const float* x = (const float*)d_in[0];
  const float* w = (const float*)d_in[1];
  float* out     = (float*)d_out;
  dim3 grid(32 * 112);   // one block per (n, h) output row
  dim3 block(TPB);
  hipLaunchKernelGGL(MBMconv2d_75917841924742_kernel, grid, block, 0, stream,
                     x, w, out);
}
